// Model_13503377178786
// MI455X (gfx1250) — matrix operand order repaired and checked by reading, not compiled, not run on hardware
//
#include <hip/hip_runtime.h>
#include <hip/hip_bf16.h>

// ---------------------------------------------------------------------------
// Model dims (compile-time constants from the reference)
// ---------------------------------------------------------------------------
#define MODEL_V  8192
#define MODEL_D  512
#define MODEL_L  2048
#define MODEL_H  8
#define MODEL_NL 6
#define MODEL_B  4
#define MODEL_HD 64
#define MODEL_M  (MODEL_B * MODEL_L)   // 8192 token rows

typedef __bf16 v16bf __attribute__((ext_vector_type(16)));
typedef float  v8f   __attribute__((ext_vector_type(8)));
typedef int    v4i   __attribute__((ext_vector_type(4)));

#define DEV static __device__ __forceinline__

// ---- gfx1250 async global->LDS copies (guarded; fall back to sync path) ----
#if defined(__AMDGCN__) && \
    __has_builtin(__builtin_amdgcn_global_load_async_to_lds_b128) && \
    __has_builtin(__builtin_amdgcn_s_wait_asynccnt)
#define USE_ASYNC_LDS 1
#else
#define USE_ASYNC_LDS 0
#endif

#if USE_ASYNC_LDS
// Builtin prototype (from clang diagnostics): pointee type is v4i, with the
// source in the global AS(1) and the destination in LDS AS(3).
typedef __attribute__((address_space(1))) v4i* glb_p4;
typedef __attribute__((address_space(3))) v4i* lds_p4;
// 16B per lane: LDS[lds] = MEM[g]
DEV void async_cp16(const void* g, void* lds) {
    // Generic->AS casts via integers: global addresses are identical in the
    // generic and global address spaces; the low 32 bits of a generic LDS
    // address are the LDS offset.
    __builtin_amdgcn_global_load_async_to_lds_b128(
        (glb_p4)(unsigned long long)g,
        (lds_p4)(unsigned int)(unsigned long long)lds,
        0, 0);
}
DEV void async_wait0() { __builtin_amdgcn_s_wait_asynccnt(0); }
#endif

// float -> bf16 (round to nearest even)
DEV unsigned short f2bf(float f) {
    union { float f; unsigned int u; } x; x.f = f;
    unsigned int u = x.u;
    u += 0x7fffu + ((u >> 16) & 1u);
    return (unsigned short)(u >> 16);
}

union FragU { uint4 q[2]; v16bf v; };

// A-operand fragment (16xK=32 bf16): lane m = lane&15 holds row m; element e:
//   e in [0,8)  -> k = hi*8 + e ;  e in [8,16) -> k = 16 + hi*8 + (e-8)
DEV v16bf load_fragA(const unsigned short* base, int row0, int pitch) {
    const int lane = threadIdx.x & 31;
    const unsigned short* r =
        base + (size_t)(row0 + (lane & 15)) * pitch + ((lane >> 4) * 8);
    FragU f;
    f.q[0] = *(const uint4*)(r);
    f.q[1] = *(const uint4*)(r + 16);
    return f.v;
}

// B-operand fragment (K=32 x 16): lane n = lane&15; element e:
//   e in [0,8)  -> k = hi*8 + e ;  e in [8,16) -> k = 16 + hi*8 + (e-8)
DEV v16bf load_fragB(const unsigned short* base, int row0, int pitch) {
    const int lane = threadIdx.x & 31;
    const unsigned short* r =
        base + (size_t)(row0 + (lane & 15)) * pitch + ((lane >> 4) * 8);
    FragU f;
    f.q[0] = *(const uint4*)(r);
    f.q[1] = *(const uint4*)(r + 16);
    return f.v;
}

DEV v8f wmma_bf16(v16bf a, v16bf b, v8f c) {
    return __builtin_amdgcn_wmma_f32_16x16x32_bf16(
        false, a, false, b, (short)0, c, false, false);
}

// ---- 16-lane reductions: DPP16 row_ror (VALU-only) with shfl fallback -----
#if defined(__AMDGCN__) && __has_builtin(__builtin_amdgcn_update_dpp)
#define DPP_ROR(x, n) \
    __builtin_amdgcn_update_dpp(0, (x), 0x120 + (n), 0xf, 0xf, true)
DEV float red_max16(float v) {
    union { float f; int i; } a, b; a.f = v;
    b.i = DPP_ROR(a.i, 8); a.f = fmaxf(a.f, b.f);
    b.i = DPP_ROR(a.i, 4); a.f = fmaxf(a.f, b.f);
    b.i = DPP_ROR(a.i, 2); a.f = fmaxf(a.f, b.f);
    b.i = DPP_ROR(a.i, 1); a.f = fmaxf(a.f, b.f);
    return a.f;
}
DEV float red_sum16(float v) {
    union { float f; int i; } a, b; a.f = v;
    b.i = DPP_ROR(a.i, 8); a.f += b.f;
    b.i = DPP_ROR(a.i, 4); a.f += b.f;
    b.i = DPP_ROR(a.i, 2); a.f += b.f;
    b.i = DPP_ROR(a.i, 1); a.f += b.f;
    return a.f;
}
#else
DEV float red_max16(float v) {
    #pragma unroll
    for (int m = 1; m < 16; m <<= 1) v = fmaxf(v, __shfl_xor(v, m));
    return v;
}
DEV float red_sum16(float v) {
    #pragma unroll
    for (int m = 1; m < 16; m <<= 1) v += __shfl_xor(v, m);
    return v;
}
#endif

// ---------------------------------------------------------------------------
// GEMM: C[M,N] = act[M,K] (bf16) @ W[N,K]^T (bf16), fp32 accum, epilogues:
//   EPI 0: out bf16 (+bias)     EPI 1: relu -> bf16 (+bias)
//   EPI 2: fp32 residual +=     EPI 3: out fp32 (+bias)
// Block 256 thr (8 waves), tile 128x128x32, wave = 32x64 (2x4 wmma frags)
// ---------------------------------------------------------------------------
template <int EPI>
__global__ __launch_bounds__(256)
void gemm_bf16_kernel(const unsigned short* __restrict__ A,
                      const unsigned short* __restrict__ Bw,
                      const float* __restrict__ bias,
                      void* __restrict__ Cout,
                      int M, int N, int K) {
    __shared__ alignas(16) unsigned short As[128 * 40];
    __shared__ alignas(16) unsigned short Bs[128 * 40];

    const int tid  = threadIdx.x;
    const int lane = tid & 31;
    const int w    = tid >> 5;      // wave 0..7
    const int wm   = w & 3;         // 4 waves along M
    const int wn   = w >> 2;        // 2 waves along N
    const int m0   = blockIdx.y * 128;
    const int n0   = blockIdx.x * 128;

    const v8f vzero = {0.f, 0.f, 0.f, 0.f, 0.f, 0.f, 0.f, 0.f};
    v8f acc[2][4];
    #pragma unroll
    for (int i = 0; i < 2; ++i)
        #pragma unroll
        for (int j = 0; j < 4; ++j) acc[i][j] = vzero;

    const int ksteps = K >> 5;
    for (int kt = 0; kt < ksteps; ++kt) {
        const int kb = kt * 32;
        #pragma unroll
        for (int p = 0; p < 2; ++p) {
            const int e   = (tid + p * 256) * 8;   // bf16 element index in tile
            const int row = e >> 5;
            const int col = e & 31;
            const unsigned short* ga = A + (size_t)(m0 + row) * K + kb + col;
            const unsigned short* gb = Bw + (size_t)(n0 + row) * K + kb + col;
#if USE_ASYNC_LDS
            async_cp16(ga, &As[row * 40 + col]);
            async_cp16(gb, &Bs[row * 40 + col]);
#else
            *(uint4*)&As[row * 40 + col] = *(const uint4*)ga;
            *(uint4*)&Bs[row * 40 + col] = *(const uint4*)gb;
#endif
            if (kt + 1 < ksteps) {   // prefetch next K tile
                __builtin_prefetch(ga + 32, 0, 1);
                __builtin_prefetch(gb + 32, 0, 1);
            }
        }
#if USE_ASYNC_LDS
        async_wait0();
#endif
        __syncthreads();

        v16bf af[2], bfg[4];
        #pragma unroll
        for (int i = 0; i < 2; ++i) af[i] = load_fragA(As, wm * 32 + i * 16, 40);
        #pragma unroll
        for (int j = 0; j < 4; ++j) bfg[j] = load_fragB(Bs, wn * 64 + j * 16, 40);
        #pragma unroll
        for (int i = 0; i < 2; ++i)
            #pragma unroll
            for (int j = 0; j < 4; ++j)
                acc[i][j] = wmma_bf16(af[i], bfg[j], acc[i][j]);
        __syncthreads();
    }

    const int hi = lane >> 4, ln = lane & 15;
    #pragma unroll
    for (int i = 0; i < 2; ++i)
        #pragma unroll
        for (int j = 0; j < 4; ++j)
            #pragma unroll
            for (int e = 0; e < 8; ++e) {
                const int m = m0 + wm * 32 + i * 16 + hi * 8 + e;
                const int n = n0 + wn * 64 + j * 16 + ln;
                float val = acc[i][j][e] + (bias ? bias[n] : 0.f);
                const size_t idx = (size_t)m * N + n;
                if (EPI == 0) {
                    ((unsigned short*)Cout)[idx] = f2bf(val);
                } else if (EPI == 1) {
                    ((unsigned short*)Cout)[idx] = f2bf(fmaxf(val, 0.f));
                } else if (EPI == 2) {
                    float* C = (float*)Cout;
                    C[idx] = C[idx] + val;
                } else {
                    ((float*)Cout)[idx] = val;
                }
            }
}

// ---------------------------------------------------------------------------
// Flash attention (causal). Grid (L/128, H, B), block 256 = 8 waves.
// Wave handles 16 query rows; block streams 32-key tiles through LDS.
// q,k,v,z are bf16 [B*L, D] with head h in columns h*64 .. h*64+63.
// ---------------------------------------------------------------------------
__global__ __launch_bounds__(256)
void attn_kernel(const unsigned short* __restrict__ q,
                 const unsigned short* __restrict__ k,
                 const unsigned short* __restrict__ v,
                 unsigned short* __restrict__ z) {
    __shared__ alignas(16) unsigned short Ks[32 * 72];        // K tile rows
    __shared__ alignas(16) unsigned short VT[64 * 40];        // V tile transposed
    __shared__ alignas(16) unsigned short Pb[8 * 16 * 40];    // per-wave P tile

    const int tid  = threadIdx.x;
    const int lane = tid & 31;
    const int w    = tid >> 5;
    const int hi   = lane >> 4, ln = lane & 15;
    const int qt   = blockIdx.x;
    const int hh   = blockIdx.y;
    const int bb   = blockIdx.z;
    const size_t mbase = (size_t)bb * MODEL_L;
    const int qr   = qt * 128 + w * 16;   // first query row (sequence-local)

    const v8f vzero = {0.f, 0.f, 0.f, 0.f, 0.f, 0.f, 0.f, 0.f};

    // Q fragments (rows qr..qr+15, K split 0-31 / 32-63)
    v16bf aq[2];
    #pragma unroll
    for (int kc = 0; kc < 2; ++kc)
        aq[kc] = load_fragA(q + (mbase + qr) * MODEL_D + hh * 64 + kc * 32, 0,
                            MODEL_D);

    v8f o[4];
    float mrow[8], lrow[8];
    #pragma unroll
    for (int of = 0; of < 4; ++of) o[of] = vzero;
    #pragma unroll
    for (int e = 0; e < 8; ++e) { mrow[e] = -1e30f; lrow[e] = 0.f; }

    const int ntiles = qt * 4 + 4;        // keys 0 .. qt*128+127 in tiles of 32
    for (int jt = 0; jt < ntiles; ++jt) {
        const int kb = jt * 32;
        {   // cooperative stage: K rows as-is (async), V transposed (scatter)
            const int row = tid >> 3;
            const int c0  = (tid & 7) * 8;
            const unsigned short* gk =
                k + (mbase + kb + row) * MODEL_D + hh * 64 + c0;
#if USE_ASYNC_LDS
            async_cp16(gk, &Ks[row * 72 + c0]);
#else
            *(uint4*)&Ks[row * 72 + c0] = *(const uint4*)gk;
#endif
            union { uint4 u; unsigned short s[8]; } t;
            t.u = *(const uint4*)(v + (mbase + kb + row) * MODEL_D + hh * 64 + c0);
            #pragma unroll
            for (int j = 0; j < 8; ++j) VT[(c0 + j) * 40 + row] = t.s[j];
        }
#if USE_ASYNC_LDS
        async_wait0();
#endif
        __syncthreads();

        // S = Q @ K^T  (16 x 32 keys -> two 16x16 C frags)
        v8f s[2] = {vzero, vzero};
        #pragma unroll
        for (int nf = 0; nf < 2; ++nf) {
            v16bf bk0 = load_fragB(Ks + 0, nf * 16, 72);
            v16bf bk1 = load_fragB(Ks + 32, nf * 16, 72);
            s[nf] = wmma_bf16(aq[0], bk0, s[nf]);
            s[nf] = wmma_bf16(aq[1], bk1, s[nf]);
        }

        // online softmax
        float p0[8], p1[8], alpha[8];
        #pragma unroll
        for (int e = 0; e < 8; ++e) {
            const int rrow = qr + e + hi * 8;
            float s0 = s[0][e] * 0.125f;
            float s1 = s[1][e] * 0.125f;
            if (kb + ln > rrow)       s0 = -1e30f;
            if (kb + 16 + ln > rrow)  s1 = -1e30f;
            float mv = red_max16(fmaxf(s0, s1));
            const float nm = fmaxf(mrow[e], mv);
            alpha[e] = __expf(mrow[e] - nm);
            mrow[e]  = nm;
            p0[e] = __expf(s0 - nm);
            p1[e] = __expf(s1 - nm);
            lrow[e] = lrow[e] * alpha[e] + red_sum16(p0[e] + p1[e]);
        }

        // P -> LDS (C-layout store), reload as A-operand fragment
        unsigned short* pw = Pb + w * 640;
        #pragma unroll
        for (int e = 0; e < 8; ++e) {
            pw[(e + hi * 8) * 40 + ln]      = f2bf(p0[e]);
            pw[(e + hi * 8) * 40 + 16 + ln] = f2bf(p1[e]);
        }
        __syncthreads();
        v16bf ap = load_fragA(pw, 0, 40);

        // O = O*alpha + P @ V
        #pragma unroll
        for (int of = 0; of < 4; ++of) {
            v16bf bv = load_fragB(VT, of * 16, 40);
            #pragma unroll
            for (int e = 0; e < 8; ++e) o[of][e] *= alpha[e];
            o[of] = wmma_bf16(ap, bv, o[of]);
        }
        __syncthreads();
    }

    #pragma unroll
    for (int of = 0; of < 4; ++of)
        #pragma unroll
        for (int e = 0; e < 8; ++e) {
            const size_t row = mbase + qr + e + hi * 8;
            const int col = hh * 64 + of * 16 + ln;
            z[row * MODEL_D + col] = f2bf(o[of][e] / lrow[e]);
        }
}

// ---------------------------------------------------------------------------
// LayerNorm: one wave32 per row of 512; bf16 output.
// ---------------------------------------------------------------------------
__global__ __launch_bounds__(256)
void layernorm_kernel(const float* __restrict__ h, const float* __restrict__ g,
                      const float* __restrict__ b,
                      unsigned short* __restrict__ out) {
    const int lane = threadIdx.x & 31;
    const size_t row = (size_t)blockIdx.x * 8 + (threadIdx.x >> 5);
    const float* hr = h + row * MODEL_D;
    float vv[16], s = 0.f, s2 = 0.f;
    #pragma unroll
    for (int j = 0; j < 16; ++j) {
        const float x = hr[lane + j * 32];
        vv[j] = x; s += x; s2 += x * x;
    }
    #pragma unroll
    for (int m = 1; m < 32; m <<= 1) { s += __shfl_xor(s, m); s2 += __shfl_xor(s2, m); }
    const float mu   = s * (1.f / MODEL_D);
    const float var  = s2 * (1.f / MODEL_D) - mu * mu;
    const float rstd = rsqrtf(var + 1e-5f);
    #pragma unroll
    for (int j = 0; j < 16; ++j) {
        const int d = lane + j * 32;
        out[row * MODEL_D + d] = f2bf((vv[j] - mu) * rstd * g[d] + b[d]);
    }
}

// ---------------------------------------------------------------------------
// Embedding: h[row] = tok_emb[x[row]] + pos_emb[row % L]
// ---------------------------------------------------------------------------
__global__ __launch_bounds__(128)
void embed_kernel(const int* __restrict__ x, const float* __restrict__ tok,
                  const float* __restrict__ pos, float* __restrict__ h) {
    const size_t row = blockIdx.x;
    const int l = (int)(row & (MODEL_L - 1));
    const int t = x[row];
    #pragma unroll
    for (int j = 0; j < 4; ++j) {
        const int d = threadIdx.x + j * 128;
        h[row * MODEL_D + d] =
            tok[(size_t)t * MODEL_D + d] + pos[(size_t)l * MODEL_D + d];
    }
}

// fp32 -> bf16 bulk conversion (grid-stride)
__global__ __launch_bounds__(256)
void cvt_bf16_kernel(const float* __restrict__ in,
                     unsigned short* __restrict__ out, size_t n) {
    size_t i = (size_t)blockIdx.x * 256 + threadIdx.x;
    const size_t stride = (size_t)gridDim.x * 256;
    for (; i < n; i += stride) out[i] = f2bf(in[i]);
}

// per-row: loss_r = logsumexp(logits[r]) - logits[r][y[r]]
__global__ __launch_bounds__(256)
void loss_row_kernel(const float* __restrict__ logits, const int* __restrict__ y,
                     float* __restrict__ lossr) {
    __shared__ float red[256];
    const int tid = threadIdx.x;
    const size_t row = blockIdx.x;
    const float* lr = logits + row * MODEL_V;
    float mx = -3.0e38f;
    for (int j = tid; j < MODEL_V; j += 256) mx = fmaxf(mx, lr[j]);
    red[tid] = mx; __syncthreads();
    for (int s = 128; s > 0; s >>= 1) {
        if (tid < s) red[tid] = fmaxf(red[tid], red[tid + s]);
        __syncthreads();
    }
    mx = red[0]; __syncthreads();
    float sm = 0.f;
    for (int j = tid; j < MODEL_V; j += 256) sm += __expf(lr[j] - mx);
    red[tid] = sm; __syncthreads();
    for (int s = 128; s > 0; s >>= 1) {
        if (tid < s) red[tid] += red[tid + s];
        __syncthreads();
    }
    if (tid == 0) lossr[row] = (mx + logf(red[0])) - lr[y[row]];
}

__global__ __launch_bounds__(256)
void loss_final_kernel(const float* __restrict__ lossr, float* __restrict__ out) {
    __shared__ float red[256];
    const int tid = threadIdx.x;
    float s = 0.f;
    for (int j = tid; j < MODEL_M; j += 256) s += lossr[j];
    red[tid] = s; __syncthreads();
    for (int st = 128; st > 0; st >>= 1) {
        if (tid < st) red[tid] += red[tid + st];
        __syncthreads();
    }
    if (tid == 0) {
        const float loss = red[0] * (1.f / MODEL_M);
        out[0] = loss;
        out[1] = __expf(loss);
    }
}

// ---------------------------------------------------------------------------
// Host-side orchestration
// ---------------------------------------------------------------------------
extern "C" void kernel_launch(void* const* d_in, const int* in_sizes, int n_in,
                              void* d_out, int out_size, void* d_ws,
                              size_t ws_size, hipStream_t stream) {
    (void)in_sizes; (void)n_in; (void)out_size; (void)ws_size;

    const int*   x    = (const int*)d_in[0];
    const int*   y    = (const int*)d_in[1];
    const float* tok  = (const float*)d_in[2];
    const float* pos  = (const float*)d_in[3];
    const float* Wq   = (const float*)d_in[4];
    const float* Wk   = (const float*)d_in[5];
    const float* Wv   = (const float*)d_in[6];
    const float* W0   = (const float*)d_in[7];
    const float* b0   = (const float*)d_in[8];
    const float* g1   = (const float*)d_in[9];
    const float* bt1  = (const float*)d_in[10];
    const float* W1   = (const float*)d_in[11];
    const float* b1   = (const float*)d_in[12];
    const float* W2   = (const float*)d_in[13];
    const float* b2   = (const float*)d_in[14];
    const float* g2   = (const float*)d_in[15];
    const float* bt2  = (const float*)d_in[16];
    const float* gf   = (const float*)d_in[17];
    const float* btf  = (const float*)d_in[18];
    const float* Wp   = (const float*)d_in[19];
    const float* bp   = (const float*)d_in[20];

    const int M = MODEL_M, D = MODEL_D, V = MODEL_V;
    const size_t DD = (size_t)D * D;

    char* ws = (char*)d_ws;
    size_t off = 0;
    auto alloc = [&](size_t bytes) -> char* {
        char* p = ws + off;
        off = (off + bytes + 255) & ~(size_t)255;
        return p;
    };
    float*          h    = (float*)alloc((size_t)M * D * 4);
    unsigned short* nbuf = (unsigned short*)alloc((size_t)M * D * 2);
    unsigned short* qb   = (unsigned short*)alloc((size_t)M * D * 2);
    unsigned short* kb   = (unsigned short*)alloc((size_t)M * D * 2);
    unsigned short* vb   = (unsigned short*)alloc((size_t)M * D * 2);
    unsigned short* zb   = (unsigned short*)alloc((size_t)M * D * 2);
    unsigned short* fb   = (unsigned short*)alloc((size_t)M * 4 * D * 2);
    unsigned short* wbuf = (unsigned short*)alloc((size_t)V * D * 2);  // >= 12*DD
    float*          lossr = (float*)alloc((size_t)M * 4);

    const dim3 gQKV(D / 128, M / 128);       // 4 x 64
    const dim3 gFF1(4 * D / 128, M / 128);   // 16 x 64
    const dim3 gLog(V / 128, M / 128);       // 64 x 64
    const dim3 gAttn(MODEL_L / 128, MODEL_H, MODEL_B);

    embed_kernel<<<M, 128, 0, stream>>>(x, tok, pos, h);

    for (int l = 0; l < MODEL_NL; ++l) {
        unsigned short* wq = wbuf;
        unsigned short* wk = wbuf + DD;
        unsigned short* wv = wbuf + 2 * DD;
        unsigned short* w0 = wbuf + 3 * DD;
        unsigned short* w1 = wbuf + 4 * DD;   // 4*DD elements
        unsigned short* w2 = wbuf + 8 * DD;   // 4*DD elements

        cvt_bf16_kernel<<<1024, 256, 0, stream>>>(Wq + (size_t)l * DD, wq, DD);
        cvt_bf16_kernel<<<1024, 256, 0, stream>>>(Wk + (size_t)l * DD, wk, DD);
        cvt_bf16_kernel<<<1024, 256, 0, stream>>>(Wv + (size_t)l * DD, wv, DD);
        cvt_bf16_kernel<<<1024, 256, 0, stream>>>(W0 + (size_t)l * DD, w0, DD);
        cvt_bf16_kernel<<<4096, 256, 0, stream>>>(W1 + (size_t)l * 4 * DD, w1, 4 * DD);
        cvt_bf16_kernel<<<4096, 256, 0, stream>>>(W2 + (size_t)l * 4 * DD, w2, 4 * DD);

        // attention sub-block
        layernorm_kernel<<<M / 8, 256, 0, stream>>>(h, g1 + l * D, bt1 + l * D, nbuf);
        gemm_bf16_kernel<0><<<gQKV, 256, 0, stream>>>(nbuf, wq, nullptr, qb, M, D, D);
        gemm_bf16_kernel<0><<<gQKV, 256, 0, stream>>>(nbuf, wk, nullptr, kb, M, D, D);
        gemm_bf16_kernel<0><<<gQKV, 256, 0, stream>>>(nbuf, wv, nullptr, vb, M, D, D);
        attn_kernel<<<gAttn, 256, 0, stream>>>(qb, kb, vb, zb);
        gemm_bf16_kernel<2><<<gQKV, 256, 0, stream>>>(zb, w0, b0 + l * D, h, M, D, D);

        // FFN sub-block
        layernorm_kernel<<<M / 8, 256, 0, stream>>>(h, g2 + l * D, bt2 + l * D, nbuf);
        gemm_bf16_kernel<1><<<gFF1, 256, 0, stream>>>(nbuf, w1, b1 + (size_t)l * 4 * D,
                                                      fb, M, 4 * D, D);
        gemm_bf16_kernel<2><<<gQKV, 256, 0, stream>>>(fb, w2, b2 + l * D, h, M, D,
                                                      4 * D);
    }

    layernorm_kernel<<<M / 8, 256, 0, stream>>>(h, gf, btf, nbuf);
    cvt_bf16_kernel<<<4096, 256, 0, stream>>>(Wp, wbuf, (size_t)V * D);
    gemm_bf16_kernel<3><<<gLog, 256, 0, stream>>>(nbuf, wbuf, bp, (float*)d_out,
                                                  M, V, D);

    loss_row_kernel<<<M, 256, 0, stream>>>((const float*)d_out, y, lossr);
    loss_final_kernel<<<1, 256, 0, stream>>>(lossr,
                                             (float*)d_out + (size_t)M * V);
}
